// ToyTransformer_63410897158506
// MI455X (gfx1250) — matrix operand order repaired and checked by reading, not compiled, not run on hardware
//
#include <hip/hip_runtime.h>
#include <hip/hip_bf16.h>

typedef __attribute__((ext_vector_type(16))) _Float16 v16h;
typedef __attribute__((ext_vector_type(8)))  float    v8f;
typedef __attribute__((ext_vector_type(4)))  unsigned int u32x4;
typedef __attribute__((ext_vector_type(8)))  int      i32x8;
typedef __attribute__((ext_vector_type(4)))  int      i32x4;

#define L_    4
#define B_    8
#define S_    1024
#define DIN_  256
#define H_    512
#define P_    1024
#define NH_   8
#define HD_   64
#define DOUT_ 256
#define M_    (B_ * S_)

// LDS weight-tile row stride in floats: 32 data + 4 pad (TDM pad feature)
#define WROW_ 36

// ---------------------------------------------------------------------------
// WMMA helper: D = A(16x32 f16) x B(32x16 f16) + C(16x16 f32)
// ---------------------------------------------------------------------------
__device__ __forceinline__ v8f wmma_f16(v16h a, v16h b, v8f c) {
  return __builtin_amdgcn_wmma_f32_16x16x32_f16(
      /*neg_a=*/false, a, /*neg_b=*/false, b,
      /*c_mod=*/(short)0, c, /*reuse_a=*/false, /*reuse_b=*/false);
}

// A-operand fragment (16x32 f16) from row-major f32 matrix (rows = M dim).
// A layout (contiguous): lane<16 holds K k0..k0+15; lane>=16 holds k0+16..k0+31.
__device__ __forceinline__ v16h frag_a_rm(const float* __restrict__ Mp,
                                          int row0, int k0, int ld) {
  const int l = threadIdx.x & 31;
  const float* p = Mp + (long)(row0 + (l & 15)) * ld + k0 + ((l & 16) ? 16 : 0);
  v16h f;
#pragma unroll
  for (int i = 0; i < 8; ++i) f[i] = (_Float16)p[i];
#pragma unroll
  for (int i = 0; i < 8; ++i) f[i + 8] = (_Float16)p[i + 8];
  return f;
}

// A-operand fragment where the logical A is the transpose of a row-major
// source: element (m,k) = Mp[k*ld + (col0+m)]. Used for V^T in attention.
__device__ __forceinline__ v16h frag_a_cm(const float* __restrict__ Mp,
                                          int col0, int k0, int ld) {
  const int l = threadIdx.x & 31;
  const int col = col0 + (l & 15);
  const int kb  = k0 + ((l & 16) ? 16 : 0);
  v16h f;
#pragma unroll
  for (int i = 0; i < 8; ++i) f[i]     = (_Float16)Mp[(long)(kb + i) * ld + col];
#pragma unroll
  for (int i = 0; i < 8; ++i) f[i + 8] = (_Float16)Mp[(long)(kb + 8 + i) * ld + col];
  return f;
}

// B-operand fragment (32x16 f16) loaded from B^T stored row-major [N x K].
// B layout (contiguous): lane n holds col n with K = k0+0..15; lane n+16 holds
// K = k0+16..31 (contiguous 16-float read).
__device__ __forceinline__ v16h frag_b_rm(const float* __restrict__ Mp,
                                          int n0, int k0, int ld) {
  const int l = threadIdx.x & 31;
  const float* p = Mp + (long)(n0 + (l & 15)) * ld + k0 + ((l & 16) ? 16 : 0);
  v16h f;
#pragma unroll
  for (int i = 0; i < 16; ++i) f[i] = (_Float16)p[i];
  return f;
}

// B-operand fragment from the LDS-staged weight tile (rows padded to WROW_
// floats by the TDM pad feature -> conflict-free & 16B-aligned reads).
__device__ __forceinline__ v16h frag_b_lds(const float* sm, int n0) {
  const int l = threadIdx.x & 31;
  const float* p = sm + (n0 + (l & 15)) * WROW_ + ((l & 16) ? 16 : 0);
  v16h f;
#pragma unroll
  for (int i = 0; i < 16; ++i) f[i] = (_Float16)p[i];
  return f;
}

// ---------------------------------------------------------------------------
// TDM: DMA a 64-row x 32-col f32 tile of W[N,K] (row-major) into LDS.
// D# per CDNA5 ISA:
//   g0: count=1 | lds_addr | global_addr | type=2
//   g1: data_size=4B, pad_enable, pad_interval=32dw(code 4), pad_amount=4dw
//       (code 3), tensor_dim0=K, tensor_dim1=0xFFFF, tile_dim0=32,
//       tile_dim1=64, tensor_dim0_stride=K
//   remaining groups: zero (2-D tensor; tile_dim2/3/4 = 0 -> unused)
// ---------------------------------------------------------------------------
__device__ __forceinline__ void tdm_load_w_tile(const float* gptr,
                                                unsigned lds_byte_off, int K) {
  const unsigned long long ga = (unsigned long long)(uintptr_t)gptr;
  u32x4 g0;
  g0[0] = 1u;                                   // count=1, is_restore=0
  g0[1] = lds_byte_off;                         // lds_addr
  g0[2] = (unsigned)(ga & 0xFFFFFFFFull);       // global_addr[31:0]
  g0[3] = (unsigned)(ga >> 32) | (2u << 30);    // global_addr[56:32] | type=2

  i32x8 g1;
  g1[0] = (int)((2u << 16) | (1u << 20) | (4u << 22) | (3u << 25));
  g1[1] = (int)(((unsigned)K & 0xFFFFu) << 16);             // tensor_dim0 lo16
  g1[2] = (int)((((unsigned)K >> 16) & 0xFFFFu) | (0xFFFFu << 16)); // dim0 hi | dim1 lo
  g1[3] = (int)(32u << 16);                                 // dim1 hi=0 | tile_dim0=32
  g1[4] = 64;                                               // tile_dim1=64, tile_dim2=0
  g1[5] = K;                                                // tensor_dim0_stride lo32
  g1[6] = 0;
  g1[7] = 0;

  const i32x4 z4 = {0, 0, 0, 0};
  const i32x8 z8 = {0, 0, 0, 0, 0, 0, 0, 0};
  __builtin_amdgcn_tensor_load_to_lds(g0, g1, z4, z4, z8, 0);
}

// ---------------------------------------------------------------------------
// GEMM with fused epilogue:  C[M,N] = A[M,K] @ W[N,K]^T + bias
//   op 0: + bias        op 1: (+bias)*mask[row]     op 2: (resid+..)*mask[row]
// Block = 256 threads (8 waves); block tile 128x64; wave tile 16x64.
// Weight tile staged in LDS by TDM, double-buffered against TENSORcnt.
// ---------------------------------------------------------------------------
__global__ __launch_bounds__(256) void gemm_ep_kernel(
    const float* __restrict__ A, const float* __restrict__ W,
    const float* __restrict__ bias, const float* __restrict__ resid,
    const unsigned char* __restrict__ mask, float* __restrict__ C,
    int M, int N, int K, int op) {
  __shared__ float wtile[2][64 * WROW_];

  const int wave = threadIdx.x >> 5;
  const int l    = threadIdx.x & 31;
  const int row0 = blockIdx.y * 128 + wave * 16;
  const int col0 = blockIdx.x * 64;
  const bool issuer = (threadIdx.x < 32);

  const unsigned lds0 = (unsigned)(uintptr_t)&wtile[0][0];
  const unsigned lds1 = (unsigned)(uintptr_t)&wtile[1][0];

  if (issuer) tdm_load_w_tile(W + (long)col0 * K, lds0, K);

  v8f acc[4] = {{0.f}, {0.f}, {0.f}, {0.f}};
  const int tiles = K >> 5;
  for (int it = 0; it < tiles; ++it) {
    if (issuer) __builtin_amdgcn_s_wait_tensorcnt(0);  // tile `it` landed
    __syncthreads();                                   // visible to all waves
    if (issuer && (it + 1) < tiles)                    // overlap next DMA
      tdm_load_w_tile(W + (long)col0 * K + (it + 1) * 32,
                      ((it + 1) & 1) ? lds1 : lds0, K);

    const float* sm = &wtile[it & 1][0];
    const int k0 = it * 32;
    const v16h a  = frag_a_rm(A, row0, k0, K);
    const v16h b0 = frag_b_lds(sm,  0);
    const v16h b1 = frag_b_lds(sm, 16);
    const v16h b2 = frag_b_lds(sm, 32);
    const v16h b3 = frag_b_lds(sm, 48);
    acc[0] = wmma_f16(a, b0, acc[0]);
    acc[1] = wmma_f16(a, b1, acc[1]);
    acc[2] = wmma_f16(a, b2, acc[2]);
    acc[3] = wmma_f16(a, b3, acc[3]);
    __syncthreads();  // done reading buf[it&1] before tile it+2 overwrites it
  }

  const int rbase = row0 + ((l & 16) ? 8 : 0);
#pragma unroll
  for (int n = 0; n < 4; ++n) {
    const int col = col0 + n * 16 + (l & 15);
    const float bv = bias[col];
#pragma unroll
    for (int r = 0; r < 8; ++r) {
      const int row = rbase + r;
      float v = acc[n][r] + bv;
      if (op >= 2) v += resid[(long)row * N + col];
      if (op >= 1) v = mask[row] ? v : 0.0f;
      C[(long)row * N + col] = v;
    }
  }
}

// ---------------------------------------------------------------------------
// LayerNorm over last dim (H_), one block (256 threads) per row.
// ---------------------------------------------------------------------------
__global__ __launch_bounds__(256) void ln_kernel(
    const float* __restrict__ X, const float* __restrict__ g,
    const float* __restrict__ b, float* __restrict__ Y) {
  __shared__ float red[256];
  const long row = blockIdx.x;
  const float* x = X + row * H_;

  float s = 0.f;
  for (int i = threadIdx.x; i < H_; i += 256) s += x[i];
  red[threadIdx.x] = s;
  __syncthreads();
  for (int off = 128; off > 0; off >>= 1) {
    if ((int)threadIdx.x < off) red[threadIdx.x] += red[threadIdx.x + off];
    __syncthreads();
  }
  const float mean = red[0] * (1.0f / H_);
  __syncthreads();

  float vs = 0.f;
  for (int i = threadIdx.x; i < H_; i += 256) {
    const float d = x[i] - mean;
    vs += d * d;
  }
  red[threadIdx.x] = vs;
  __syncthreads();
  for (int off = 128; off > 0; off >>= 1) {
    if ((int)threadIdx.x < off) red[threadIdx.x] += red[threadIdx.x + off];
    __syncthreads();
  }
  const float rstd = rsqrtf(red[0] * (1.0f / H_) + 1e-5f);

  float* y = Y + row * H_;
  for (int i = threadIdx.x; i < H_; i += 256)
    y[i] = (x[i] - mean) * rstd * g[i] + b[i];
}

// ---------------------------------------------------------------------------
// SiLU-gate:  act[m,p] = silu(up[m,p]) * up[m,P+p]
// ---------------------------------------------------------------------------
__global__ void silu_gate_kernel(const float* __restrict__ U,
                                 float* __restrict__ Y) {
  const long i = (long)blockIdx.x * blockDim.x + threadIdx.x;
  const long total = (long)M_ * P_;
  if (i >= total) return;
  const long m = i / P_;
  const int  p = (int)(i % P_);
  const float up   = U[m * (2L * P_) + p];
  const float gate = U[m * (2L * P_) + P_ + p];
  Y[i] = (up / (1.0f + __expf(-up))) * gate;
}

// ---------------------------------------------------------------------------
// Flash-style attention. Grid (S/64, NH, B); block 128 = 4 waves; each wave
// owns 16 queries and streams 32-key tiles. scores^T keeps keys in the WMMA
// M dim so softmax reductions are lane-local + one shfl_xor(16); P is rebuilt
// as a B-operand directly from the score accumulators; O^T += V^T @ P.
// ---------------------------------------------------------------------------
__global__ __launch_bounds__(128) void attn_kernel(
    const float* __restrict__ QKV, const unsigned char* __restrict__ mask,
    float* __restrict__ Out) {
  const int b    = blockIdx.z;
  const int head = blockIdx.y;
  const int wave = threadIdx.x >> 5;
  const int l    = threadIdx.x & 31;
  const int q0   = blockIdx.x * 64 + wave * 16;
  const int ld   = 3 * H_;

  const float* base = QKV + (long)b * S_ * ld;
  const float* Qp = base + head * HD_;
  const float* Kp = base + H_ + head * HD_;
  const float* Vp = base + 2 * H_ + head * HD_;
  const unsigned char* mrow = mask + b * S_;

  const v16h qf0 = frag_b_rm(Qp, q0,  0, ld);
  const v16h qf1 = frag_b_rm(Qp, q0, 32, ld);

  v8f o0 = {0.f}, o1 = {0.f}, o2 = {0.f}, o3 = {0.f};
  float mcur = -__builtin_inff();
  float lsum = 0.f;
  const int kb = (l & 16) ? 8 : 0;

  for (int kt = 0; kt < S_; kt += 32) {
    v8f c0 = {0.f}, c1 = {0.f};
    c0 = wmma_f16(frag_a_rm(Kp, kt,       0, ld), qf0, c0);
    c0 = wmma_f16(frag_a_rm(Kp, kt,      32, ld), qf1, c0);
    c1 = wmma_f16(frag_a_rm(Kp, kt + 16,  0, ld), qf0, c1);
    c1 = wmma_f16(frag_a_rm(Kp, kt + 16, 32, ld), qf1, c1);

    float tmax = -__builtin_inff();
#pragma unroll
    for (int r = 0; r < 8; ++r) {
      const float s0 = mrow[kt + kb + r]      ? c0[r] * 0.125f : -__builtin_inff();
      const float s1 = mrow[kt + 16 + kb + r] ? c1[r] * 0.125f : -__builtin_inff();
      c0[r] = s0; c1[r] = s1;
      tmax = fmaxf(tmax, fmaxf(s0, s1));
    }
    tmax = fmaxf(tmax, __shfl_xor(tmax, 16, 32));
    const float mnew  = fmaxf(mcur, tmax);
    const float alpha = (mcur == -__builtin_inff()) ? 0.f : __expf(mcur - mnew);
    mcur = mnew;

    float e0[8], e1[8];
    float psum = 0.f;
#pragma unroll
    for (int r = 0; r < 8; ++r) {
      e0[r] = __expf(c0[r] - mnew);
      e1[r] = __expf(c1[r] - mnew);
      psum += e0[r] + e1[r];
    }
    psum += __shfl_xor(psum, 16, 32);
    lsum = lsum * alpha + psum;

    v16h pf;
#pragma unroll
    for (int r = 0; r < 8; ++r) {
      const float x0 = __shfl_xor(e0[r], 16, 32);
      const float x1 = __shfl_xor(e1[r], 16, 32);
      pf[r]     = (_Float16)((l & 16) ? x1    : e0[r]);
      pf[r + 8] = (_Float16)((l & 16) ? e1[r] : x0);
    }

#pragma unroll
    for (int r = 0; r < 8; ++r) {
      o0[r] *= alpha; o1[r] *= alpha; o2[r] *= alpha; o3[r] *= alpha;
    }
    o0 = wmma_f16(frag_a_cm(Vp,  0, kt, ld), pf, o0);
    o1 = wmma_f16(frag_a_cm(Vp, 16, kt, ld), pf, o1);
    o2 = wmma_f16(frag_a_cm(Vp, 32, kt, ld), pf, o2);
    o3 = wmma_f16(frag_a_cm(Vp, 48, kt, ld), pf, o3);
  }

  const float inv = 1.0f / lsum;
  const int q = q0 + (l & 15);
  float* orow = Out + ((long)b * S_ + q) * H_ + head * HD_;
  const v8f os[4] = {o0, o1, o2, o3};
#pragma unroll
  for (int t = 0; t < 4; ++t)
#pragma unroll
    for (int r = 0; r < 8; ++r)
      orow[t * 16 + r + kb] = os[t][r] * inv;
}

// ---------------------------------------------------------------------------
// Launcher
// ---------------------------------------------------------------------------
extern "C" void kernel_launch(void* const* d_in, const int* in_sizes, int n_in,
                              void* d_out, int out_size, void* d_ws, size_t ws_size,
                              hipStream_t stream) {
  (void)in_sizes; (void)n_in; (void)out_size; (void)ws_size;

  const float*         x    = (const float*)d_in[0];
  const unsigned char* mask = (const unsigned char*)d_in[1];  // jax bool = 1B
  /* d_in[2] active_rows: unused in dense path */
  const float* in_w   = (const float*)d_in[3];
  const float* in_b   = (const float*)d_in[4];
  const float* ln1_g  = (const float*)d_in[5];
  const float* ln1_b  = (const float*)d_in[6];
  const float* qkv_w  = (const float*)d_in[7];
  const float* qkv_b  = (const float*)d_in[8];
  const float* out_w  = (const float*)d_in[9];
  const float* out_b  = (const float*)d_in[10];
  const float* ln2_g  = (const float*)d_in[11];
  const float* ln2_b  = (const float*)d_in[12];
  const float* up_w   = (const float*)d_in[13];
  const float* up_b   = (const float*)d_in[14];
  const float* down_w = (const float*)d_in[15];
  const float* down_b = (const float*)d_in[16];
  const float* outp_w = (const float*)d_in[17];
  const float* outp_b = (const float*)d_in[18];
  float* out = (float*)d_out;

  float* ws   = (float*)d_ws;
  float* h    = ws; ws += (long)M_ * H_;
  float* y    = ws; ws += (long)M_ * H_;
  float* qkv  = ws; ws += (long)M_ * 3 * H_;
  float* attn = ws; ws += (long)M_ * H_;
  float* up   = ws; ws += (long)M_ * 2 * P_;
  float* act  = ws; ws += (long)M_ * P_;

  auto gemm = [&](const float* A, const float* W, const float* bias,
                  const float* resid, float* C, int N, int K, int op) {
    dim3 grid(N / 64, M_ / 128);
    gemm_ep_kernel<<<grid, dim3(256), 0, stream>>>(A, W, bias, resid, mask, C,
                                                   M_, N, K, op);
  };

  // h = (x @ in_w^T + in_b) * maskf
  gemm(x, in_w, in_b, nullptr, h, H_, DIN_, 1);

  for (int i = 0; i < L_; ++i) {
    ln_kernel<<<M_, 256, 0, stream>>>(h, ln1_g + i * H_, ln1_b + i * H_, y);
    gemm(y, qkv_w + (long)i * 3 * H_ * H_, qkv_b + (long)i * 3 * H_,
         nullptr, qkv, 3 * H_, H_, 0);
    attn_kernel<<<dim3(S_ / 64, NH_, B_), dim3(128), 0, stream>>>(qkv, mask, attn);
    gemm(attn, out_w + (long)i * H_ * H_, out_b + (long)i * H_, h, h, H_, H_, 2);

    ln_kernel<<<M_, 256, 0, stream>>>(h, ln2_g + i * H_, ln2_b + i * H_, y);
    gemm(y, up_w + (long)i * 2 * P_ * H_, up_b + (long)i * 2 * P_,
         nullptr, up, 2 * P_, H_, 0);
    {
      const long total = (long)M_ * P_;
      silu_gate_kernel<<<(unsigned)((total + 255) / 256), 256, 0, stream>>>(up, act);
    }
    gemm(act, down_w + (long)i * H_ * P_, down_b + (long)i * H_, h, h, H_, P_, 2);
  }

  // out = (h @ outp_w^T + outp_b) * maskf
  gemm(h, outp_w, outp_b, nullptr, out, DOUT_, H_, 1);
}
